// SSD_11982958756200
// MI455X (gfx1250) — compile-verified, matrix operand order checked by reading, not run on hardware
//
#include <hip/hip_runtime.h>
#include <hip/hip_bf16.h>
#include <stdint.h>

typedef _Float16 half_t;
typedef _Float16 v16h __attribute__((ext_vector_type(16)));
typedef float    v8f  __attribute__((ext_vector_type(8)));
typedef int      v4i  __attribute__((ext_vector_type(4)));

#define TM 128          // output positions per block tile
#define TN 128          // output channels per block tile
#define TK 32           // K step; every Cin is a multiple of 32 so K tiles are contiguous
#define LDSS 40         // padded LDS row stride in halfs (80B rows, 16B-aligned chunks)

// CDNA5 async global->LDS path (ASYNCcnt); guarded so either toolchain compiles
#if defined(__gfx1250__) && __has_builtin(__builtin_amdgcn_global_load_async_to_lds_b128)
#define HAVE_ASYNC_LDS 1
#define GLOBAL_AS __attribute__((address_space(1)))
#define LDS_AS    __attribute__((address_space(3)))
#define COPY16(gp, lp) __builtin_amdgcn_global_load_async_to_lds_b128( \
    (GLOBAL_AS v4i*)(gp), (LDS_AS v4i*)(lp), 0, 0)
#if __has_builtin(__builtin_amdgcn_s_wait_asynccnt)
#define WAIT_ASYNC() __builtin_amdgcn_s_wait_asynccnt(0)
#else
#define WAIT_ASYNC() asm volatile("s_wait_asynccnt 0x0" ::: "memory")
#endif
#else
#define COPY16(gp, lp) (*(uint4*)(lp) = *(const uint4*)(gp))
#define WAIT_ASYNC()
#endif

// ---------------- f32 -> f16 (flat copy; used for 1x1 weights) ----------------
__global__ void cvt_f32_f16_kernel(const float* __restrict__ s, half_t* __restrict__ d, int n) {
    int i = blockIdx.x * blockDim.x + threadIdx.x;
    if (i < n) d[i] = (half_t)s[i];
}

// ---------------- 3x3 weight pack: OIHW f32 -> [Cout][kh][kw][ci] f16 ----------------
__global__ void pack_w3_kernel(const float* __restrict__ s, half_t* __restrict__ d,
                               int Cin, int total) {
    int i = blockIdx.x * blockDim.x + threadIdx.x;
    if (i >= total) return;
    int cin9 = 9 * Cin;
    int o    = i / cin9;
    int r    = i - o * cin9;
    int sp   = r / Cin;          // kh*3+kw
    int ci   = r - sp * Cin;
    d[i] = (half_t)s[(size_t)(o * Cin + ci) * 9 + sp];
}

// ---------------- NCHW f32 -> NHWC f16 ----------------
__global__ void nchw2nhwc_kernel(const float* __restrict__ s, half_t* __restrict__ d,
                                 int C, int HW, int total) {
    int i = blockIdx.x * blockDim.x + threadIdx.x;
    if (i >= total) return;
    int c = i % C;
    int t = i / C;
    int p = t % HW;
    int n = t / HW;
    d[i] = (half_t)s[((size_t)n * C + c) * HW + p];
}

// ---------------- L2Norm(feat0) -> NHWC f16 ----------------
__global__ void l2norm_kernel(const float* __restrict__ x, const float* __restrict__ sc,
                              half_t* __restrict__ y, int C, int HW, int NP) {
    int p = blockIdx.x * blockDim.x + threadIdx.x;
    if (p >= NP) return;
    int n = p / HW, hw = p - n * HW;
    const float* b = x + (size_t)n * C * HW + hw;
    float s = 0.f;
    for (int c = 0; c < C; ++c) { float v = b[(size_t)c * HW]; s = fmaf(v, v, s); }
    float inv = 1.f / (sqrtf(s) + 1e-10f);
    half_t* o = y + (size_t)p * C;                    // NHWC: contiguous channels
    for (int c = 0; c < C; ++c) o[c] = (half_t)(sc[c] * b[(size_t)c * HW] * inv);
}

// ---------------- implicit-GEMM conv via v_wmma_f32_16x16x32_f16 ----------------
// X: NHWC f16, Wt: [Cout][kh][kw][ci] f16; K-order = (kh,kw,ci) => contiguous 32-K tiles
// MODE 0: f16 NHWC + ReLU to workspace (extra layers)
// MODE 1: scatter conf into d_out (N,8732,85)[...,0:81]
// MODE 2: scatter loc  into d_out (N,8732,85)[...,81:85]
template <int KS, int MODE>
__global__ __launch_bounds__(256)
void conv_wmma_kernel(const half_t* __restrict__ X, const half_t* __restrict__ Wt,
                      const float* __restrict__ bias,
                      half_t* __restrict__ Yh, float* __restrict__ Yf,
                      int NBATCH, int Cin, int H, int W, int Ho, int Wo, int Cout,
                      int stride, int pad, int K, int abase, int mbox)
{
    __shared__ __align__(16) half_t As[2][TM * LDSS];
    __shared__ __align__(16) half_t Bs[2][TN * LDSS];

    const int tid    = threadIdx.x;
    const int lane   = tid & 31;
    const int wave   = tid >> 5;           // 0..7
    const int wave_m = (wave & 3) * 32;    // 4 M-strips of 32
    const int wave_n = (wave >> 2) * 64;   // 2 N-strips of 64
    const int M      = NBATCH * Ho * Wo;
    const int m0     = blockIdx.x * TM;
    const int n0     = blockIdx.y * TN;

    // fill assignment: 16B chunk (tid&3)*8 of rows (tid>>2) and (tid>>2)+64
    const int frow = tid >> 2;
    const int coff = (tid & 3) * 8;

    // ---- prologue: hoist all row decompositions out of the K loop ----
    int  pbase[2], hb[2], wb[2]; bool mok[2];
    #pragma unroll
    for (int i = 0; i < 2; ++i) {
        int m  = m0 + frow + i * 64;
        mok[i] = m < M;
        int mm = mok[i] ? m : 0;
        int n   = mm / (Ho * Wo);
        int pos = mm - n * (Ho * Wo);
        int ho  = pos / Wo;
        int wo  = pos - ho * Wo;
        pbase[i] = n * H * W;
        hb[i] = ho * stride - pad;
        wb[i] = wo * stride - pad;
    }
    const half_t* browp[2]; bool bok[2];
    #pragma unroll
    for (int i = 0; i < 2; ++i) {
        int gn  = n0 + frow + i * 64;
        bok[i]  = gn < Cout;
        browp[i] = Wt + (size_t)(bok[i] ? gn : 0) * K;
    }

    auto fillB = [&](int buf, int k0) {
        #pragma unroll
        for (int i = 0; i < 2; ++i) {
            half_t* lp = &Bs[buf][(frow + i * 64) * LDSS + coff];
            if (bok[i]) { COPY16(browp[i] + k0 + coff, lp); }
            else        { *(uint4*)lp = make_uint4(0u, 0u, 0u, 0u); }
        }
    };
    auto fillA = [&](int buf, int kh, int kw, int cio) {
        #pragma unroll
        for (int i = 0; i < 2; ++i) {
            half_t* lp = &As[buf][(frow + i * 64) * LDSS + coff];
            int hi = hb[i] + kh, wi = wb[i] + kw;
            bool ok = mok[i] && (unsigned)hi < (unsigned)H && (unsigned)wi < (unsigned)W;
            if (ok) { COPY16(&X[(size_t)(pbase[i] + hi * W + wi) * Cin + cio + coff], lp); }
            else    { *(uint4*)lp = make_uint4(0u, 0u, 0u, 0u); }
        }
    };

    v8f acc[2][4] = {};

    // incremental (kh,kw,ci) walk -> zero divisions inside the loop
    int kh = 0, kw = 0, cio = 0;
    fillB(0, 0);
    fillA(0, 0, 0, 0);
    WAIT_ASYNC();
    __syncthreads();

    const int kb    = (lane < 16) ? 0 : 8;
    const int arow0 = wave_m + (lane & 15);
    const int brow0 = wave_n + (lane & 15);
    union F { v16h v; uint4 q[2]; };

    for (int k0 = 0; k0 < K; k0 += TK) {
        const int cur = (k0 >> 5) & 1, nxt = cur ^ 1;
        // advance k-state and prefetch next tiles (overlaps with WMMA below)
        int nkh = kh, nkw = kw, ncio = cio + TK;
        if (ncio == Cin) { ncio = 0; if constexpr (KS == 3) { if (++nkw == 3) { nkw = 0; ++nkh; } } }
        if (k0 + TK < K) { fillB(nxt, k0 + TK); fillA(nxt, nkh, nkw, ncio); }
        kh = nkh; kw = nkw; cio = ncio;

        F af[2];
        #pragma unroll
        for (int mi = 0; mi < 2; ++mi) {
            const half_t* ap = &As[cur][(arow0 + mi * 16) * LDSS + kb];
            af[mi].q[0] = *(const uint4*)ap;
            af[mi].q[1] = *(const uint4*)(ap + 16);
        }
        #pragma unroll
        for (int ns = 0; ns < 4; ++ns) {
            const half_t* bp = &Bs[cur][(brow0 + ns * 16) * LDSS + kb];
            F bf;
            bf.q[0] = *(const uint4*)bp;
            bf.q[1] = *(const uint4*)(bp + 16);
            acc[0][ns] = __builtin_amdgcn_wmma_f32_16x16x32_f16(
                false, af[0].v, false, bf.v, (short)0, acc[0][ns], false, false);
            acc[1][ns] = __builtin_amdgcn_wmma_f32_16x16x32_f16(
                false, af[1].v, false, bf.v, (short)0, acc[1][ns], false, false);
        }
        WAIT_ASYNC();
        __syncthreads();
    }

    // ---- epilogue: C layout VGPR r -> M = r + 8*(lane>=16), N = lane&15 ----
    #pragma unroll
    for (int mi = 0; mi < 2; ++mi) {
        #pragma unroll
        for (int ns = 0; ns < 4; ++ns) {
            int gn = n0 + wave_n + ns * 16 + (lane & 15);
            #pragma unroll
            for (int r = 0; r < 8; ++r) {
                int gm = m0 + wave_m + mi * 16 + r + 8 * (lane >> 4);
                if (gm >= M || gn >= Cout) continue;
                float v = acc[mi][ns][r] + bias[gn];
                int n   = gm / (Ho * Wo);
                int pos = gm - n * (Ho * Wo);
                if constexpr (MODE == 0) {
                    v = v > 0.f ? v : 0.f;    // all extra layers are ReLU'd
                    Yh[((size_t)gm) * Cout + gn] = (half_t)v;   // NHWC, gm == n*HoWo+pos
                } else if constexpr (MODE == 1) {
                    int box = gn / 81, cls = gn - box * 81;
                    size_t anchor = (size_t)abase + (size_t)pos * mbox + box;
                    Yf[((size_t)n * 8732 + anchor) * 85 + cls] = v;
                } else {
                    int box = gn >> 2, coord = gn & 3;
                    size_t anchor = (size_t)abase + (size_t)pos * mbox + box;
                    Yf[((size_t)n * 8732 + anchor) * 85 + 81 + coord] = v;
                }
            }
        }
    }
}

// ------------------------------------------------------------------------
// SSD300 topology tables
static const int EX_O[8]  = {256, 512, 128, 256, 128, 256, 128, 256};
static const int EX_I[8]  = {1024, 256, 512, 128, 256, 128, 256, 128};
static const int EX_KK[8] = {1, 3, 1, 3, 1, 3, 1, 3};
static const int EX_S[8]  = {1, 2, 1, 2, 1, 1, 1, 1};
static const int EX_P[8]  = {0, 1, 0, 1, 0, 0, 0, 0};

static const int LC[6] = {512, 1024, 512, 256, 256, 256};
static const int LH[6] = {38, 19, 10, 5, 3, 1};
static const int LM[6] = {4, 6, 6, 6, 4, 4};
static const int LB[6] = {0, 5776, 7942, 8542, 8692, 8728};

static void launch_conv(hipStream_t stream,
                        const half_t* X, const half_t* Wt, const float* bias,
                        half_t* Yh, float* Yf,
                        int NB, int Cin, int H, int Ho, int Cout,
                        int ks, int s, int p, int mode, int abase, int mbox) {
    int K = Cin * ks * ks;
    int M = NB * Ho * Ho;
    dim3 grid((M + TM - 1) / TM, (Cout + TN - 1) / TN);
    if (mode == 0) {
        if (ks == 1)
            conv_wmma_kernel<1, 0><<<grid, 256, 0, stream>>>(X, Wt, bias, Yh, Yf,
                NB, Cin, H, H, Ho, Ho, Cout, s, p, K, abase, mbox);
        else
            conv_wmma_kernel<3, 0><<<grid, 256, 0, stream>>>(X, Wt, bias, Yh, Yf,
                NB, Cin, H, H, Ho, Ho, Cout, s, p, K, abase, mbox);
    } else if (mode == 1) {
        conv_wmma_kernel<3, 1><<<grid, 256, 0, stream>>>(X, Wt, bias, Yh, Yf,
            NB, Cin, H, H, Ho, Ho, Cout, s, p, K, abase, mbox);
    } else {
        conv_wmma_kernel<3, 2><<<grid, 256, 0, stream>>>(X, Wt, bias, Yh, Yf,
            NB, Cin, H, H, Ho, Ho, Cout, s, p, K, abase, mbox);
    }
}

extern "C" void kernel_launch(void* const* d_in, const int* in_sizes, int n_in,
                              void* d_out, int out_size, void* d_ws, size_t ws_size,
                              hipStream_t stream) {
    (void)in_sizes; (void)n_in; (void)out_size; (void)ws_size;
    const int NB = 16;
    const float* feat0 = (const float*)d_in[0];
    const float* feat1 = (const float*)d_in[1];
    const float* l2w   = (const float*)d_in[2];
    float* out = (float*)d_out;

    // bump allocator over workspace (~70 MB of f16 staging)
    char* ws = (char*)d_ws;
    size_t cur = 0;
    auto alloc_h = [&](size_t nelem) -> half_t* {
        half_t* p = (half_t*)(ws + cur);
        cur = (cur + nelem * sizeof(half_t) + 255) & ~(size_t)255;
        return p;
    };
    auto cvt = [&](const float* src, half_t* dst, int n) {
        cvt_f32_f16_kernel<<<(n + 255) / 256, 256, 0, stream>>>(src, dst, n);
    };
    auto pack3 = [&](const float* src, half_t* dst, int Cin, int total) {
        pack_w3_kernel<<<(total + 255) / 256, 256, 0, stream>>>(src, dst, Cin, total);
    };

    half_t* f0h = alloc_h((size_t)NB * 512 * 38 * 38);    // NHWC
    half_t* f1h = alloc_h((size_t)NB * 1024 * 19 * 19);   // NHWC

    // pack extra-layer weights, allocate extra-layer outputs (NHWC)
    half_t* ewp[8]; half_t* eout[8];
    int Hs[9]; Hs[0] = 19;
    for (int i = 0; i < 8; ++i) {
        int Ho = (Hs[i] + 2 * EX_P[i] - EX_KK[i]) / EX_S[i] + 1;
        Hs[i + 1] = Ho;
        int wn = EX_O[i] * EX_I[i] * EX_KK[i] * EX_KK[i];
        ewp[i] = alloc_h((size_t)wn);
        if (EX_KK[i] == 1) cvt((const float*)d_in[3 + 2 * i], ewp[i], wn);
        else               pack3((const float*)d_in[3 + 2 * i], ewp[i], EX_I[i], wn);
        eout[i] = alloc_h((size_t)NB * EX_O[i] * Ho * Ho);
    }
    // pack head weights (all 3x3)
    half_t* cwp[6]; half_t* bwp[6];
    for (int i = 0; i < 6; ++i) {
        int cn = LM[i] * 81 * LC[i] * 9;
        int bn = LM[i] * 4  * LC[i] * 9;
        cwp[i] = alloc_h((size_t)cn); pack3((const float*)d_in[19 + 4 * i], cwp[i], LC[i], cn);
        bwp[i] = alloc_h((size_t)bn); pack3((const float*)d_in[21 + 4 * i], bwp[i], LC[i], bn);
    }

    // activation prep (NHWC f16)
    {
        int total = NB * 1024 * 19 * 19;
        nchw2nhwc_kernel<<<(total + 255) / 256, 256, 0, stream>>>(feat1, f1h, 1024, 19 * 19, total);
        int NP = NB * 38 * 38;
        l2norm_kernel<<<(NP + 255) / 256, 256, 0, stream>>>(feat0, l2w, f0h, 512, 38 * 38, NP);
    }

    // extra-layer chain (ReLU, NHWC f16 outputs)
    const half_t* xin = f1h;
    int Hin = 19;
    for (int i = 0; i < 8; ++i) {
        int Ho = Hs[i + 1];
        launch_conv(stream, xin, ewp[i], (const float*)d_in[4 + 2 * i], eout[i], nullptr,
                    NB, EX_I[i], Hin, Ho, EX_O[i], EX_KK[i], EX_S[i], EX_P[i],
                    /*mode=*/0, 0, 0);
        xin = eout[i]; Hin = Ho;
    }

    // detection heads: scatter directly into packed (N,8732,85) f32 output
    const half_t* feats[6] = {f0h, f1h, eout[1], eout[3], eout[5], eout[7]};
    for (int i = 0; i < 6; ++i) {
        launch_conv(stream, feats[i], cwp[i], (const float*)d_in[20 + 4 * i], nullptr, out,
                    NB, LC[i], LH[i], LH[i], LM[i] * 81, 3, 1, 1, /*mode=*/1, LB[i], LM[i]);
        launch_conv(stream, feats[i], bwp[i], (const float*)d_in[22 + 4 * i], nullptr, out,
                    NB, LC[i], LH[i], LH[i], LM[i] * 4, 3, 1, 1, /*mode=*/2, LB[i], LM[i]);
    }
}
